// Block_58497454571919
// MI455X (gfx1250) — matrix operand order repaired and checked by reading, not compiled, not run on hardware
//
#include <hip/hip_runtime.h>
#include <hip/hip_bf16.h>

typedef __bf16          bf16x16 __attribute__((ext_vector_type(16)));
typedef float           f32x8   __attribute__((ext_vector_type(8)));
typedef unsigned short  u16x8   __attribute__((ext_vector_type(8)));

#define NB      4
#define SEQ     2048
#define DMODEL  2048
#define NHEAD   16
#define DHEAD   128
#define DFF     8192
#define ROWS    (NB * SEQ)            // 8192
#define NTOT    (ROWS * DMODEL)       // 16,777,216

// ---------------------------------------------------------------- helpers
__device__ __forceinline__ unsigned short f2bf(float f) {
  unsigned u = __float_as_uint(f);
  u += 0x7fffu + ((u >> 16) & 1u);   // round-to-nearest-even
  return (unsigned short)(u >> 16);
}

union FragU { u16x8 h[2]; bf16x16 v; };

// A-operand 16x32 bf16 fragment (ISA 7.12.2): lane row = lane&15;
// lanes 0-15 hold K {0..7, 16..23}; lanes 16-31 hold K {8..15, 24..31}.
__device__ __forceinline__ bf16x16 ldsfragA(const unsigned short* prow, int lane) {
  int k0 = (lane & 16) ? 8 : 0;
  FragU u;
  u.h[0] = *(const u16x8*)(prow + k0);
  u.h[1] = *(const u16x8*)(prow + k0 + 16);
  return u.v;
}

// B-operand 32x16 bf16 fragment: lane col = lane&15;
// lanes 0-15 hold K {0..7, 16..23}; lanes 16-31 hold K {8..15, 24..31}.
__device__ __forceinline__ bf16x16 ldsfragB(const unsigned short* prow, int lane) {
  int k0 = (lane & 16) ? 8 : 0;
  FragU u;
  u.h[0] = *(const u16x8*)(prow + k0);
  u.h[1] = *(const u16x8*)(prow + k0 + 16);
  return u.v;
}

__device__ __forceinline__ f32x8 zero8() {
  f32x8 z = {0.f,0.f,0.f,0.f,0.f,0.f,0.f,0.f};
  return z;
}

__device__ __forceinline__ float gelu_tanh(float x) {
  return 0.5f * x * (1.f + tanhf(0.7978845608028654f * (x + 0.044715f * x * x * x)));
}

// CDNA5 async global->LDS copy (ASYNCcnt-tracked, no VGPR data path).
// LDS dest operand = wave-relative LDS byte offset = low 32 bits of the
// generic pointer (aperture sits in the high bits).
__device__ __forceinline__ void async_load_b128(void* lds, const void* gptr) {
  asm volatile("global_load_async_to_lds_b128 %0, %1, off"
               :: "v"((unsigned)(unsigned long long)lds), "v"(gptr)
               : "memory");
}
__device__ __forceinline__ void wait_async0() {
  asm volatile("s_wait_asynccnt 0x0" ::: "memory");
}

// ---------------------------------------------------------------- fp32 -> bf16
__global__ __launch_bounds__(256) void k_cvt(const float* __restrict__ in,
                                             unsigned short* __restrict__ out, int n) {
  int stride = gridDim.x * blockDim.x;
  for (int i = blockIdx.x * blockDim.x + threadIdx.x; i < n; i += stride)
    out[i] = f2bf(in[i]);
}

// ---------------------------------------------------------------- bf16 GEMM
// C[M,N] = A[M,K] * Bt[N,K]^T + bias[N]   (A, Bt row-major bf16; acc fp32)
// Workgroup tile 128x128, BK=32, 8 waves (4x2), wave tile 32x64 = 2x4 WMMA tiles.
// Double-buffered LDS staging via global_load_async_to_lds_b128 (1 barrier/iter).
template<int STORE_F32, int STORE_BF16, int GELU>
__global__ __launch_bounds__(256)
void k_gemm(const unsigned short* __restrict__ A, int lda,
            const unsigned short* __restrict__ Bt, int ldb,
            const float* __restrict__ bias,
            float* Cf, unsigned short* Cb,
            int N, int K) {
  __shared__ __align__(16) unsigned short sA[2][128 * 40];
  __shared__ __align__(16) unsigned short sB[2][128 * 40];

  const int tid  = threadIdx.x;
  const int lane = tid & 31;
  const int wv   = tid >> 5;
  const int rl   = lane & 15;
  const int m0w  = (wv & 3) * 32;
  const int n0w  = (wv >> 2) * 64;
  const int mblk = blockIdx.y * 128;
  const int nblk = blockIdx.x * 128;

  f32x8 acc[2][4];
#pragma unroll
  for (int mi = 0; mi < 2; ++mi)
#pragma unroll
    for (int ni = 0; ni < 4; ++ni) acc[mi][ni] = zero8();

  // issue async staging for K-chunk kk into buffer `buf`
  auto issue = [&](int buf, int kk) {
#pragma unroll
    for (int it = 0; it < 2; ++it) {
      int s   = tid + it * 256;
      int r   = s >> 2;
      int off = (s & 3) * 8;
      async_load_b128(&sA[buf][r * 40 + off], &A[(mblk + r) * lda + kk + off]);
      async_load_b128(&sB[buf][r * 40 + off], &Bt[(nblk + r) * ldb + kk + off]);
    }
  };

  issue(0, 0);
  int ib = 0;
  for (int kk = 0; kk < K; kk += 32, ib ^= 1) {
    wait_async0();       // this wave's async writes into sA/sB[ib] are in LDS
    __syncthreads();     // ...and every other wave's too; prior reads of ib^1 done
    if (kk + 32 < K) issue(ib ^ 1, kk + 32);

    bf16x16 af[2], bfr[4];
#pragma unroll
    for (int mi = 0; mi < 2; ++mi)
      af[mi] = ldsfragA(&sA[ib][(m0w + mi * 16 + rl) * 40], lane);
#pragma unroll
    for (int ni = 0; ni < 4; ++ni)
      bfr[ni] = ldsfragB(&sB[ib][(n0w + ni * 16 + rl) * 40], lane);
#pragma unroll
    for (int mi = 0; mi < 2; ++mi)
#pragma unroll
      for (int ni = 0; ni < 4; ++ni)
        acc[mi][ni] = __builtin_amdgcn_wmma_f32_16x16x32_bf16(
            false, af[mi], false, bfr[ni], (short)0, acc[mi][ni], false, false);
  }

  const int hlf = (lane & 16) ? 8 : 0;
#pragma unroll
  for (int mi = 0; mi < 2; ++mi)
#pragma unroll
    for (int ni = 0; ni < 4; ++ni) {
      int nglob = nblk + n0w + ni * 16 + rl;
      float bs = bias[nglob];
#pragma unroll
      for (int r = 0; r < 8; ++r) {
        int mglob = mblk + m0w + mi * 16 + r + hlf;
        float v = acc[mi][ni][r] + bs;
        if (GELU) v = gelu_tanh(v);
        if (STORE_F32)  Cf[mglob * N + nglob] = v;
        if (STORE_BF16) Cb[mglob * N + nglob] = f2bf(v);
      }
    }
}

// ---------------------------------------------------------------- fused attention
// One workgroup per (b,h). Scores S = Qh^T Kh / sqrt(S) (128x128, contract over
// seq), softmax rows, Y_h = P * Vh^T -> Y[b, h*128+d, s] (reference's reshape).
__global__ __launch_bounds__(256)
void k_attn(const unsigned short* __restrict__ QKV, unsigned short* __restrict__ Y) {
  __shared__ __align__(16) unsigned char smem[135680];
  unsigned short* sP = (unsigned short*)smem;              // [128][136] bf16
  unsigned short* sV = (unsigned short*)(smem + 34816);    // [128][136] bf16
  float*          sS = (float*)(smem + 69632);             // [128][129] f32
  unsigned short* sQ = (unsigned short*)(smem + 69632);    // [128][40] (overlaps sS)
  unsigned short* sK = (unsigned short*)(smem + 69632 + 10240);

  const int blk = blockIdx.x;
  const int b = blk >> 4, h = blk & 15;
  const int qoff = h * 128, koff = 2048 + h * 128, voff = 4096 + h * 128;
  const int rowbase = b * SEQ;

  const int tid  = threadIdx.x;
  const int lane = tid & 31;
  const int wv   = tid >> 5;
  const int rl   = lane & 15;
  const int m0w  = (wv & 3) * 32;
  const int n0w  = (wv >> 2) * 64;
  const int hlf  = (lane & 16) ? 8 : 0;

  // ---- phase 1: scores ----
  f32x8 acc[2][4];
#pragma unroll
  for (int mi = 0; mi < 2; ++mi)
#pragma unroll
    for (int ni = 0; ni < 4; ++ni) acc[mi][ni] = zero8();

  const int sl   = tid & 31;         // s within chunk
  const int dseg = (tid >> 5) * 16;  // 16 head-channels per thread

  for (int kk = 0; kk < SEQ; kk += 32) {
    const unsigned short* gq = &QKV[(rowbase + kk + sl) * 6144 + qoff + dseg];
    const unsigned short* gk = &QKV[(rowbase + kk + sl) * 6144 + koff + dseg];
    u16x8 q0 = *(const u16x8*)gq, q1 = *(const u16x8*)(gq + 8);
    u16x8 c0 = *(const u16x8*)gk, c1 = *(const u16x8*)(gk + 8);
#pragma unroll
    for (int j = 0; j < 8; ++j) {
      sQ[(dseg + j) * 40 + sl]     = q0[j];
      sQ[(dseg + 8 + j) * 40 + sl] = q1[j];
      sK[(dseg + j) * 40 + sl]     = c0[j];
      sK[(dseg + 8 + j) * 40 + sl] = c1[j];
    }
    __syncthreads();

    bf16x16 af[2], bfr[4];
#pragma unroll
    for (int mi = 0; mi < 2; ++mi)
      af[mi] = ldsfragA(&sQ[(m0w + mi * 16 + rl) * 40], lane);
#pragma unroll
    for (int ni = 0; ni < 4; ++ni)
      bfr[ni] = ldsfragB(&sK[(n0w + ni * 16 + rl) * 40], lane);
#pragma unroll
    for (int mi = 0; mi < 2; ++mi)
#pragma unroll
      for (int ni = 0; ni < 4; ++ni)
        acc[mi][ni] = __builtin_amdgcn_wmma_f32_16x16x32_bf16(
            false, af[mi], false, bfr[ni], (short)0, acc[mi][ni], false, false);
    __syncthreads();
  }

  const float rs = 0.02209708691207961f;  // 1/sqrt(2048)
#pragma unroll
  for (int mi = 0; mi < 2; ++mi)
#pragma unroll
    for (int ni = 0; ni < 4; ++ni)
#pragma unroll
      for (int r = 0; r < 8; ++r)
        sS[(m0w + mi * 16 + r + hlf) * 129 + n0w + ni * 16 + rl] = acc[mi][ni][r] * rs;
  __syncthreads();

  // ---- phase 2: row softmax -> sP (bf16) ----
  if (tid < 128) {
    float mx = -3.0e38f;
    for (int e = 0; e < 128; ++e) mx = fmaxf(mx, sS[tid * 129 + e]);
    float sm = 0.f;
    for (int e = 0; e < 128; ++e) {
      float ev = expf(sS[tid * 129 + e] - mx);
      sS[tid * 129 + e] = ev;
      sm += ev;
    }
    float inv = 1.f / sm;
    for (int e = 0; e < 128; ++e) sP[tid * 136 + e] = f2bf(sS[tid * 129 + e] * inv);
  }
  __syncthreads();

  // ---- phase 3: Y_h = P * Vh^T, tiles of 128 output columns ----
  for (int ns = 0; ns < SEQ; ns += 128) {
    {
      int row = tid >> 1, half = (tid & 1) * 64;
      const unsigned short* gv = &QKV[(rowbase + ns + row) * 6144 + voff + half];
#pragma unroll
      for (int j = 0; j < 64; j += 8)
        async_load_b128(&sV[row * 136 + half + j], gv + j);
      wait_async0();
    }
    __syncthreads();

    f32x8 yac[2][4];
#pragma unroll
    for (int mi = 0; mi < 2; ++mi)
#pragma unroll
      for (int ni = 0; ni < 4; ++ni) yac[mi][ni] = zero8();

#pragma unroll
    for (int kc = 0; kc < 128; kc += 32) {
      bf16x16 af[2], bfr[4];
#pragma unroll
      for (int mi = 0; mi < 2; ++mi)
        af[mi] = ldsfragA(&sP[(m0w + mi * 16 + rl) * 136 + kc], lane);
#pragma unroll
      for (int ni = 0; ni < 4; ++ni)
        bfr[ni] = ldsfragB(&sV[(n0w + ni * 16 + rl) * 136 + kc], lane);
#pragma unroll
      for (int mi = 0; mi < 2; ++mi)
#pragma unroll
        for (int ni = 0; ni < 4; ++ni)
          yac[mi][ni] = __builtin_amdgcn_wmma_f32_16x16x32_bf16(
              false, af[mi], false, bfr[ni], (short)0, yac[mi][ni], false, false);
    }

#pragma unroll
    for (int mi = 0; mi < 2; ++mi)
#pragma unroll
      for (int ni = 0; ni < 4; ++ni) {
        int col = ns + n0w + ni * 16 + rl;
#pragma unroll
        for (int r = 0; r < 8; ++r) {
          int drow = m0w + mi * 16 + r + hlf;
          Y[(rowbase + h * 128 + drow) * DMODEL + col] = f2bf(yac[mi][ni][r]);
        }
      }
    __syncthreads();
  }
}

// ---------------------------------------------------------------- global LN
__global__ void k_zero2(double* p) { if (threadIdx.x < 2) p[threadIdx.x] = 0.0; }

__global__ __launch_bounds__(256)
void k_ln_reduce(const float* a, const float* __restrict__ x, float* t,
                 double* acc, int n) {
  __shared__ float rs[256], rq[256];
  float s = 0.f, q = 0.f;
  for (int i = blockIdx.x * 256 + threadIdx.x; i < n; i += gridDim.x * 256) {
    float v = a[i] + x[i];
    t[i] = v;
    s += v;
    q += v * v;
  }
  rs[threadIdx.x] = s; rq[threadIdx.x] = q;
  __syncthreads();
  for (int o = 128; o > 0; o >>= 1) {
    if (threadIdx.x < o) {
      rs[threadIdx.x] += rs[threadIdx.x + o];
      rq[threadIdx.x] += rq[threadIdx.x + o];
    }
    __syncthreads();
  }
  if (threadIdx.x == 0) {
    atomicAdd(&acc[0], (double)rs[0]);
    atomicAdd(&acc[1], (double)rq[0]);
  }
}

__global__ __launch_bounds__(256)
void k_ln_norm(const float* __restrict__ t, const double* __restrict__ acc,
               const float* __restrict__ w, const float* __restrict__ bb,
               float* __restrict__ yf, unsigned short* yb, int n) {
  double mean = acc[0] / (double)n;
  double var  = (acc[1] - (double)n * mean * mean) / (double)(n - 1);
  float inv = (float)(1.0 / sqrt(var + 1e-12));
  float mf  = (float)mean;
  for (int i = blockIdx.x * 256 + threadIdx.x; i < n; i += gridDim.x * 256) {
    int c = i & (DMODEL - 1);
    float v = (t[i] - mf) * inv * w[c] + bb[c];
    yf[i] = v;
    if (yb) yb[i] = f2bf(v);
  }
}

// ---------------------------------------------------------------- launch
extern "C" void kernel_launch(void* const* d_in, const int* in_sizes, int n_in,
                              void* d_out, int out_size, void* d_ws, size_t ws_size,
                              hipStream_t stream) {
  (void)in_sizes; (void)n_in; (void)out_size; (void)ws_size;
  const float* x      = (const float*)d_in[0];
  const float* W1w    = (const float*)d_in[1];
  const float* W1bias = (const float*)d_in[2];
  const float* W2w    = (const float*)d_in[3];
  const float* W2bias = (const float*)d_in[4];
  const float* fcw    = (const float*)d_in[5];
  const float* fcbias = (const float*)d_in[6];
  const float* pjw    = (const float*)d_in[7];
  const float* pjbias = (const float*)d_in[8];
  const float* ln1w   = (const float*)d_in[9];
  const float* ln1b   = (const float*)d_in[10];
  const float* ln2w   = (const float*)d_in[11];
  const float* ln2b   = (const float*)d_in[12];

  char* ws = (char*)d_ws;
  size_t o = 0;
  unsigned short* xb    = (unsigned short*)(ws + o); o += (size_t)NTOT * 2;
  unsigned short* W1b   = (unsigned short*)(ws + o); o += (size_t)3 * DMODEL * DMODEL * 2;
  unsigned short* W2b   = (unsigned short*)(ws + o); o += (size_t)DMODEL * DMODEL * 2;
  unsigned short* fcb   = (unsigned short*)(ws + o); o += (size_t)DFF * DMODEL * 2;
  unsigned short* pjb   = (unsigned short*)(ws + o); o += (size_t)DMODEL * DFF * 2;
  unsigned short* QKVb  = (unsigned short*)(ws + o); o += (size_t)ROWS * 3 * DMODEL * 2;
  unsigned short* Yb    = (unsigned short*)(ws + o); o += (size_t)NTOT * 2;
  float*          A1f   = (float*)(ws + o);          o += (size_t)NTOT * 4;
  float*          x1f   = (float*)(ws + o);          o += (size_t)NTOT * 4;
  unsigned short* x1b   = (unsigned short*)(ws + o); o += (size_t)NTOT * 2;
  unsigned short* Hb    = (unsigned short*)(ws + o); o += (size_t)ROWS * DFF * 2;
  double*         accd  = (double*)(ws + o);         o += 64;

  // fp32 -> bf16 conversions
  k_cvt<<<2048, 256, 0, stream>>>(x,   xb,  NTOT);
  k_cvt<<<2048, 256, 0, stream>>>(W1w, W1b, 3 * DMODEL * DMODEL);
  k_cvt<<<1024, 256, 0, stream>>>(W2w, W2b, DMODEL * DMODEL);
  k_cvt<<<2048, 256, 0, stream>>>(fcw, fcb, DFF * DMODEL);
  k_cvt<<<2048, 256, 0, stream>>>(pjw, pjb, DMODEL * DFF);

  // QKV = x @ W1^T + b1  -> bf16 [8192, 6144]
  k_gemm<0,1,0><<<dim3(48, 64), 256, 0, stream>>>(xb, DMODEL, W1b, DMODEL, W1bias,
                                                  nullptr, QKVb, 3 * DMODEL, DMODEL);
  // fused attention -> Yb bf16 [8192, 2048]
  k_attn<<<64, 256, 0, stream>>>(QKVb, Yb);
  // attn out = Y @ W2^T + b2 -> fp32
  k_gemm<1,0,0><<<dim3(16, 64), 256, 0, stream>>>(Yb, DMODEL, W2b, DMODEL, W2bias,
                                                  A1f, nullptr, DMODEL, DMODEL);
  // LN1 over (attn + x)
  k_zero2<<<1, 32, 0, stream>>>(accd);
  k_ln_reduce<<<2048, 256, 0, stream>>>(A1f, x, A1f, accd, NTOT);
  k_ln_norm<<<2048, 256, 0, stream>>>(A1f, accd, ln1w, ln1b, x1f, x1b, NTOT);
  // H = gelu(x1 @ fc^T + fc_b) -> bf16 [8192, 8192]
  k_gemm<0,1,1><<<dim3(64, 64), 256, 0, stream>>>(x1b, DMODEL, fcb, DMODEL, fcbias,
                                                  nullptr, Hb, DFF, DMODEL);
  // M = H @ proj^T + proj_b -> fp32
  k_gemm<1,0,0><<<dim3(16, 64), 256, 0, stream>>>(Hb, DFF, pjb, DFF, pjbias,
                                                  A1f, nullptr, DMODEL, DFF);
  // LN2 over (M + x1) -> d_out
  k_zero2<<<1, 32, 0, stream>>>(accd + 2);
  k_ln_reduce<<<2048, 256, 0, stream>>>(A1f, x1f, A1f, accd + 2, NTOT);
  k_ln_norm<<<2048, 256, 0, stream>>>(A1f, accd + 2, ln2w, ln2b,
                                      (float*)d_out, nullptr, NTOT);
}
